// MultiHeadAttention_17678085390814
// MI455X (gfx1250) — matrix operand order repaired and checked by reading, not compiled, not run on hardware
//
#include <hip/hip_runtime.h>
#include <hip/hip_bf16.h>

// ---------------------------------------------------------------------------
// MI455X (gfx1250) fused multi-head attention:
//   qkv = x @ W{q,k,v}^T  (+ fused RoPE + RMS-norm on q,k)  -> f16 [B,H,T,D]
//   flash attention (online softmax, f16 WMMA, f32 accum)   -> f16 [B,T,C]
//   out = y @ Wo^T                                          -> f32 [B,T,C]
// All matmuls via v_wmma_f32_16x16x32_f16 (wave32).
// GEMM main loop: double-buffered LDS tiles streamed with
// GLOBAL_LOAD_ASYNC_TO_LDS_B128 (ASYNCcnt pipeline).
// ---------------------------------------------------------------------------

#define N_HEAD   16
#define HEAD_DIM 64
#define N_EMBD   1024
#define BATCH    2
#define SEQ      2048
#define MTOT     (BATCH * SEQ)      // 4096 rows for projection GEMMs

typedef __attribute__((ext_vector_type(4)))  float    v4f;
typedef __attribute__((ext_vector_type(8)))  float    v8f;
typedef __attribute__((ext_vector_type(8)))  _Float16 v8h;
typedef __attribute__((ext_vector_type(16))) _Float16 v16h;
typedef int v4i_ __attribute__((__vector_size__(16)));   // builtin's pointee type

union HVec16 { v16h v; v8h h[2]; };

// Compose a 16-half WMMA operand from two 16-byte LDS chunks.
__device__ __forceinline__ v16h ld16(const _Float16* plo, const _Float16* phi) {
    HVec16 u;
    u.h[0] = *(const v8h*)plo;
    u.h[1] = *(const v8h*)phi;
    return u.v;
}

__device__ __forceinline__ v8f wmma16(v16h a, v16h b, v8f c) {
    // 8-arg form: (neg_a, A, neg_b, B, c_mod, C, reuse_a, reuse_b)
    return __builtin_amdgcn_wmma_f32_16x16x32_f16(false, a, false, b,
                                                  (short)0, c, false, false);
}

// ---- CDNA5 async global->LDS copy (512 B per op across the wave) ----------
#if defined(__HIP_DEVICE_COMPILE__) && __has_builtin(__builtin_amdgcn_global_load_async_to_lds_b128)
#define HAVE_ASYNC_LDS 1
#else
#define HAVE_ASYNC_LDS 0
#endif

__device__ __forceinline__ void async_b128(_Float16* lds, const _Float16* g) {
#if HAVE_ASYNC_LDS
    // Signature (from compiler diagnostic): (v4i addrspace(1)* gsrc,
    //                                        v4i addrspace(3)* ldst,
    //                                        imm offset, imm cpol)
    __builtin_amdgcn_global_load_async_to_lds_b128(
        (__attribute__((address_space(1))) v4i_*)(g),
        (__attribute__((address_space(3))) v4i_*)(lds),
        0, 0);
#else
    *(v8h*)lds = *(const v8h*)g;   // synchronous fallback, same 16 B granule
#endif
}

template <int N>
__device__ __forceinline__ void wait_async() {
#if defined(__HIP_DEVICE_COMPILE__) && __has_builtin(__builtin_amdgcn_s_wait_asynccnt)
    __builtin_amdgcn_s_wait_asynccnt(N);
#else
    asm volatile("s_wait_asynccnt %0" :: "i"(N) : "memory");
#endif
}

// ---------------------------------------------------------------------------
// Kernel 1: f32 -> f16 convert, 8 elems/thread (b128 in, b128 out)
// ---------------------------------------------------------------------------
__global__ __launch_bounds__(256) void cvt_f32_f16(const float* __restrict__ in,
                                                   _Float16* __restrict__ out,
                                                   int n) {
    int i = (blockIdx.x * 256 + threadIdx.x) * 8;
    if (i + 8 <= n) {
        v4f a = *(const v4f*)(in + i);
        v4f b = *(const v4f*)(in + i + 4);
        v8h o;
        #pragma unroll
        for (int j = 0; j < 4; ++j) { o[j] = (_Float16)a[j]; o[j + 4] = (_Float16)b[j]; }
        *(v8h*)(out + i) = o;
    }
}

// ---------------------------------------------------------------------------
// Kernel 2: tiled WMMA GEMM   C[m,n] = sum_k A[m,k] * W[n,k]
//   block tile 128x64, K-step 32 (native WMMA K), 8 waves = 4(M) x 2(N),
//   each wave: 2x2 16x16 accumulator tiles. Double-buffered async tiles:
//   per tile each wave issues 2 A-ops + 1 B-op (512 B each) -> wait to <=3
//   drains the previous tile (ASYNCcnt completes in order per the ISA).
// mode 0: write f32 [M][N]                         (final out-proj)
// mode 1: write f16 [B,H,T,D]                      (V)
// mode 2: RoPE + RMS-norm epilogue, f16 [B,H,T,D]  (Q, K)
// ---------------------------------------------------------------------------
union GemmSmem {
    struct {
        _Float16 As[2][128][40];  // double-buffered M x K tiles (80 B rows, 16B aligned)
        _Float16 Bs[2][64][40];   // double-buffered N x K tiles, N-major
    } ab;
    float Cs[128][66];            // f32 staging for the rowwise epilogue
};

// Issue one K-tile's async copies (3 ops per wave: 2 for A, 1 for B).
__device__ __forceinline__ void issue_tile(const _Float16* __restrict__ A,
                                           const _Float16* __restrict__ W,
                                           int K, int m0, int n0, int kk,
                                           _Float16 (*As)[40], _Float16 (*Bs)[40],
                                           int wave, int lane) {
    // A tile: 128 rows x 32 halfs = 8 KB = 16 wave-ops; 2 per wave.
    #pragma unroll
    for (int o = 0; o < 2; ++o) {
        int g   = (wave * 2 + o) * 32 + lane;    // 16-B segment index
        int row = g >> 2, seg = g & 3;
        async_b128(&As[row][seg * 8], A + (size_t)(m0 + row) * K + kk + seg * 8);
    }
    // B tile: 64 rows x 32 halfs = 4 KB = 8 wave-ops; 1 per wave.
    {
        int g   = wave * 32 + lane;
        int row = g >> 2, seg = g & 3;
        async_b128(&Bs[row][seg * 8], W + (size_t)(n0 + row) * K + kk + seg * 8);
    }
}

__global__ __launch_bounds__(256) void gemm_wmma(
    const _Float16* __restrict__ A,      // [M][K] row-major
    const _Float16* __restrict__ W,      // [N][K] row-major
    float* __restrict__ outF32,          // mode 0
    _Float16* __restrict__ outBHTD,      // mode 1/2
    const float* __restrict__ cosT,      // [T][HEAD_DIM/2]
    const float* __restrict__ sinT,
    int M, int N, int K, int mode) {

    __shared__ GemmSmem sm;

    const int tid   = threadIdx.x;
    const int wave  = tid >> 5;
    const int lane  = tid & 31;
    const int lhalf = lane >> 4;      // 0: lanes 0-15, 1: lanes 16-31
    const int l16   = lane & 15;
    const int mw    = wave >> 1;      // 0..3  (M direction)
    const int nw    = wave & 1;       // 0..1  (N direction)
    const int m0    = blockIdx.x * 128;
    const int n0    = blockIdx.y * 64;

    v8f acc[2][2] = {};

    const int NT = K / 32;
    // Prologue: stream tile 0 into buffer 0.
    issue_tile(A, W, K, m0, n0, 0, sm.ab.As[0], sm.ab.Bs[0], wave, lane);

    for (int it = 0; it < NT; ++it) {
        const int cur = it & 1;
        if (it + 1 < NT) {
            // Previous iteration's trailing barrier guarantees nobody still
            // reads buffer (1-cur); stream the next tile into it now.
            issue_tile(A, W, K, m0, n0, (it + 1) * 32,
                       sm.ab.As[1 - cur], sm.ab.Bs[1 - cur], wave, lane);
            wait_async<3>();          // drain tile `it`, keep tile `it+1` in flight
        } else {
            wait_async<0>();          // last tile: drain everything
        }
        __syncthreads();              // publish tile `it` to all waves

        // A fragment (16-bit 16x32): lanes 0-15: K 0-7 & 16-23; lanes 16-31: K 8-15 & 24-31
        const int akb = lhalf * 8;
        v16h af[2];
        #pragma unroll
        for (int i = 0; i < 2; ++i) {
            int row = mw * 32 + i * 16 + l16;
            af[i] = ld16(&sm.ab.As[cur][row][akb], &sm.ab.As[cur][row][akb + 16]);
        }
        // B fragment (32x16): lanes 0-15: K 0-7 & 16-23; lanes 16-31: K 8-15 & 24-31,
        // read from the N-major LDS tile.
        const int bkb = lhalf * 8;
        v16h bf[2];
        #pragma unroll
        for (int j = 0; j < 2; ++j) {
            int col = nw * 32 + j * 16 + l16;
            bf[j] = ld16(&sm.ab.Bs[cur][col][bkb], &sm.ab.Bs[cur][col][bkb + 16]);
        }
        #pragma unroll
        for (int i = 0; i < 2; ++i)
            #pragma unroll
            for (int j = 0; j < 2; ++j)
                acc[i][j] = wmma16(af[i], bf[j], acc[i][j]);

        __syncthreads();              // all waves done reading buffer `cur`
    }

    if (mode == 0) {
        // Direct f32 writeback from accumulator layout:
        // element (lane, v): row = v + 8*lhalf, col = l16 within the 16x16 tile.
        #pragma unroll
        for (int i = 0; i < 2; ++i)
            #pragma unroll
            for (int j = 0; j < 2; ++j)
                #pragma unroll
                for (int v = 0; v < 8; ++v) {
                    int m = m0 + mw * 32 + i * 16 + v + 8 * lhalf;
                    int c = n0 + nw * 32 + j * 16 + l16;
                    outF32[(size_t)m * N + c] = acc[i][j][v];
                }
        return;
    }

    // ---- fused epilogue for Q/K/V: stage f32 tile, rowwise RoPE + RMS-norm ----
    #pragma unroll
    for (int i = 0; i < 2; ++i)
        #pragma unroll
        for (int j = 0; j < 2; ++j)
            #pragma unroll
            for (int v = 0; v < 8; ++v)
                sm.Cs[mw * 32 + i * 16 + v + 8 * lhalf][nw * 32 + j * 16 + l16] = acc[i][j][v];
    __syncthreads();

    if (tid < 128) {
        const int r    = tid;
        const int m    = m0 + r;
        const int b    = m >> 11;        // / SEQ
        const int t    = m & (SEQ - 1);
        const int head = blockIdx.y;     // N-tile width 64 == one head

        float z[HEAD_DIM];
        if (mode == 2) {
            #pragma unroll
            for (int d = 0; d < 32; ++d) {
                float x1 = sm.Cs[r][d], x2 = sm.Cs[r][d + 32];
                float co = cosT[t * 32 + d], si = sinT[t * 32 + d];
                z[d]      =  x1 * co + x2 * si;
                z[d + 32] = -x1 * si + x2 * co;
            }
        } else {
            #pragma unroll
            for (int d = 0; d < HEAD_DIM; ++d) z[d] = sm.Cs[r][d];
        }
        float s = 1.0f;
        if (mode == 2) {
            float ms = 0.0f;
            #pragma unroll
            for (int d = 0; d < HEAD_DIM; ++d) ms += z[d] * z[d];
            ms *= (1.0f / HEAD_DIM);
            s = rsqrtf(ms + 1.1920929e-7f);   // float32 eps
        }
        _Float16* o = outBHTD + (((size_t)b * N_HEAD + head) * SEQ + t) * HEAD_DIM;
        #pragma unroll
        for (int c = 0; c < HEAD_DIM; c += 8) {
            v8h h;
            #pragma unroll
            for (int j = 0; j < 8; ++j) h[j] = (_Float16)(z[c + j] * s);
            *(v8h*)(o + c) = h;
        }
    }
}

// ---------------------------------------------------------------------------
// Kernel 3: flash attention. Block = (b, head, 128-row Q tile), 8 waves.
// Each wave owns 16 complete query rows -> online softmax reduces purely
// inside a 16-lane half-wave (shfl_xor 1/2/4/8 never crosses the half).
// ---------------------------------------------------------------------------
__global__ __launch_bounds__(256) void attn_wmma(
    const _Float16* __restrict__ Qh,   // [B,H,T,D]
    const _Float16* __restrict__ Kh,
    const _Float16* __restrict__ Vh,
    _Float16* __restrict__ Yh) {       // [B,T,C]

    __shared__ alignas(16) _Float16 Qs[128][72];
    __shared__ alignas(16) _Float16 Ks[64][72];
    __shared__ alignas(16) _Float16 Vt[64][72];     // transposed: Vt[d][j]
    __shared__ alignas(16) _Float16 Ps[8][16][72];  // per-wave P staging

    const int tid   = threadIdx.x;
    const int wave  = tid >> 5;
    const int lane  = tid & 31;
    const int lhalf = lane >> 4;
    const int l16   = lane & 15;
    const int head  = blockIdx.y;
    const int b     = blockIdx.z;
    const int t0    = blockIdx.x * 128;
    const size_t bh = ((size_t)b * N_HEAD + head) * SEQ;
    const float scale = 0.125f;   // 1/sqrt(64)

    {   // load Q tile (row-major, 64 halfs/row)
        int row = tid >> 1, seg = tid & 1;
        const _Float16* g = Qh + (bh + t0 + row) * HEAD_DIM + seg * 32;
        #pragma unroll
        for (int c = 0; c < 32; c += 8)
            *(v8h*)&Qs[row][seg * 32 + c] = *(const v8h*)(g + c);
    }

    float mrow[8], lrow[8];
    #pragma unroll
    for (int v = 0; v < 8; ++v) { mrow[v] = -__builtin_inff(); lrow[v] = 0.0f; }
    v8f oacc[4] = {};

    for (int kt = 0; kt < SEQ; kt += 64) {
        __syncthreads();
        {   // K tile row-major; V tile transposed into Vt[d][j]
            int j = tid >> 2, seg = tid & 3;
            const _Float16* gk = Kh + (bh + kt + j) * HEAD_DIM + seg * 16;
            *(v8h*)&Ks[j][seg * 16]     = *(const v8h*)gk;
            *(v8h*)&Ks[j][seg * 16 + 8] = *(const v8h*)(gk + 8);
            const _Float16* gv = Vh + (bh + kt + j) * HEAD_DIM + seg * 16;
            v8h a = *(const v8h*)gv, c2 = *(const v8h*)(gv + 8);
            #pragma unroll
            for (int e = 0; e < 8; ++e) {
                Vt[seg * 16 + e][j]     = a[e];
                Vt[seg * 16 + 8 + e][j] = c2[e];
            }
        }
        __syncthreads();

        // ---- S = Q K^T (16 x 64 per wave), d in two 32-chunks ----
        v8f sacc[4] = {};
        #pragma unroll
        for (int dc = 0; dc < 2; ++dc) {
            const int akb = lhalf * 8 + dc * 32;
            const int row = wave * 16 + l16;
            v16h af = ld16(&Qs[row][akb], &Qs[row][akb + 16]);
            const int bkb = lhalf * 8 + dc * 32;
            #pragma unroll
            for (int jt = 0; jt < 4; ++jt) {
                int col = jt * 16 + l16;
                v16h bf = ld16(&Ks[col][bkb], &Ks[col][bkb + 16]);
                sacc[jt] = wmma16(af, bf, sacc[jt]);
            }
        }

        // ---- online softmax (row = v + 8*lhalf, spread over 16-lane half) ----
        #pragma unroll
        for (int v = 0; v < 8; ++v) {
            float mx = -__builtin_inff();
            #pragma unroll
            for (int jt = 0; jt < 4; ++jt) mx = fmaxf(mx, sacc[jt][v] * scale);
            #pragma unroll
            for (int off = 1; off < 16; off <<= 1) mx = fmaxf(mx, __shfl_xor(mx, off, 32));
            float mn   = fmaxf(mrow[v], mx);
            float corr = __expf(mrow[v] - mn);
            mrow[v] = mn;
            float rs = 0.0f;
            #pragma unroll
            for (int jt = 0; jt < 4; ++jt) {
                float p = __expf(sacc[jt][v] * scale - mn);
                sacc[jt][v] = p;
                rs += p;
            }
            #pragma unroll
            for (int off = 1; off < 16; off <<= 1) rs += __shfl_xor(rs, off, 32);
            lrow[v] = lrow[v] * corr + rs;
            #pragma unroll
            for (int dt = 0; dt < 4; ++dt) oacc[dt][v] *= corr;
        }

        // ---- relayout P: accumulator layout -> A-fragment layout via LDS ----
        #pragma unroll
        for (int jt = 0; jt < 4; ++jt)
            #pragma unroll
            for (int v = 0; v < 8; ++v)
                Ps[wave][v + 8 * lhalf][jt * 16 + l16] = (_Float16)sacc[jt][v];
        asm volatile("s_wait_dscnt 0" ::: "memory");  // same-wave DS RAW fence

        // ---- O += P V  (j in two 32-chunks; Vt gives contiguous-K B frags) ----
        #pragma unroll
        for (int jc = 0; jc < 2; ++jc) {
            const int akb = lhalf * 8 + jc * 32;
            v16h af = ld16(&Ps[wave][l16][akb], &Ps[wave][l16][akb + 16]);
            const int bkb = lhalf * 8 + jc * 32;
            #pragma unroll
            for (int dt = 0; dt < 4; ++dt) {
                int col = dt * 16 + l16;
                v16h bf = ld16(&Vt[col][bkb], &Vt[col][bkb + 16]);
                oacc[dt] = wmma16(af, bf, oacc[dt]);
            }
        }
    }

    // ---- normalize and write y as f16 [B,T,C] for the out-projection GEMM ----
    #pragma unroll
    for (int dt = 0; dt < 4; ++dt)
        #pragma unroll
        for (int v = 0; v < 8; ++v) {
            int t  = t0 + wave * 16 + v + 8 * lhalf;
            float y = oacc[dt][v] / lrow[v];
            Yh[((size_t)b * SEQ + t) * N_EMBD + head * HEAD_DIM + dt * 16 + l16] = (_Float16)y;
        }
}

// ---------------------------------------------------------------------------
// Host-side orchestration
// ---------------------------------------------------------------------------
extern "C" void kernel_launch(void* const* d_in, const int* in_sizes, int n_in,
                              void* d_out, int out_size, void* d_ws, size_t ws_size,
                              hipStream_t stream) {
    const float* x    = (const float*)d_in[0];
    const float* cosp = (const float*)d_in[1];
    const float* sinp = (const float*)d_in[2];
    const float* Wq   = (const float*)d_in[3];
    const float* Wk   = (const float*)d_in[4];
    const float* Wv   = (const float*)d_in[5];
    const float* Wo   = (const float*)d_in[6];
    float* out = (float*)d_out;

    const size_t nX = (size_t)MTOT * N_EMBD;       // 4,194,304
    const size_t nW = (size_t)N_EMBD * N_EMBD;     // 1,048,576

    _Float16* ws  = (_Float16*)d_ws;
    _Float16* Xh  = ws;            ws += nX;
    _Float16* Wqh = ws;            ws += nW;
    _Float16* Wkh = ws;            ws += nW;
    _Float16* Wvh = ws;            ws += nW;
    _Float16* Woh = ws;            ws += nW;
    _Float16* Qh  = ws;            ws += nX;       // [B,H,T,D]
    _Float16* Kh  = ws;            ws += nX;
    _Float16* Vh  = ws;            ws += nX;
    _Float16* Yh  = ws;            ws += nX;       // [B,T,C]

    // 1) f32 -> f16
    cvt_f32_f16<<<dim3((unsigned)(nX / 2048)), 256, 0, stream>>>(x,  Xh,  (int)nX);
    cvt_f32_f16<<<dim3((unsigned)(nW / 2048)), 256, 0, stream>>>(Wq, Wqh, (int)nW);
    cvt_f32_f16<<<dim3((unsigned)(nW / 2048)), 256, 0, stream>>>(Wk, Wkh, (int)nW);
    cvt_f32_f16<<<dim3((unsigned)(nW / 2048)), 256, 0, stream>>>(Wv, Wvh, (int)nW);
    cvt_f32_f16<<<dim3((unsigned)(nW / 2048)), 256, 0, stream>>>(Wo, Woh, (int)nW);

    // 2) QKV projections with fused RoPE + RMS-norm epilogue
    dim3 gg(MTOT / 128, N_EMBD / 64);   // (32, 16): N-tile == one head
    gemm_wmma<<<gg, 256, 0, stream>>>(Xh, Wqh, nullptr, Qh, cosp, sinp, MTOT, N_EMBD, N_EMBD, 2);
    gemm_wmma<<<gg, 256, 0, stream>>>(Xh, Wkh, nullptr, Kh, cosp, sinp, MTOT, N_EMBD, N_EMBD, 2);
    gemm_wmma<<<gg, 256, 0, stream>>>(Xh, Wvh, nullptr, Vh, nullptr, nullptr, MTOT, N_EMBD, N_EMBD, 1);

    // 3) flash attention
    attn_wmma<<<dim3(SEQ / 128, N_HEAD, BATCH), 256, 0, stream>>>(Qh, Kh, Vh, Yh);

    // 4) output projection -> f32
    gemm_wmma<<<gg, 256, 0, stream>>>(Yh, Woh, out, nullptr, nullptr, nullptr, MTOT, N_EMBD, N_EMBD, 0);
}
